// VQElastic_26405458936344
// MI455X (gfx1250) — matrix operand order repaired and checked by reading, not compiled, not run on hardware
//
#include <hip/hip_runtime.h>
#include <hip/hip_bf16.h>

// ---------------- problem constants ----------------
#define B_   32
#define T_   512
#define E_   512           // E_DIM
#define NE_  512           // N_E
#define KK_  513           // codebook rows (N_E + 1)
#define M_   (B_ * T_)     // 16384 flattened rows of z
#define ZQSZ ((size_t)M_ * E_)   // 8,388,608 floats

#define BETA_        0.25f
#define PERS_STEP    (0.1f / 512.0f)         // PERSISTENCE / N_E
#define FLINCH_B     0.05f                   // 0.1 / (1 + 1.0)
#define EXPLORE_B    0.025f                  // 0.5 * 0.1 / (1 + 1.0)
#define EPS_         (1e-6f / 512.0f)

typedef __attribute__((ext_vector_type(16))) _Float16 v16h;
typedef __attribute__((ext_vector_type(8)))  _Float16 v8h;
typedef __attribute__((ext_vector_type(4)))  _Float16 v4h;
typedef __attribute__((ext_vector_type(8)))  float    v8f;

// ---------------- small scratch init ----------------
__global__ void init_small(float* cnt, float* ssum, float* acnt, float* lacc,
                           int* bmax, int* bmin) {
    int i = blockIdx.x * 256 + threadIdx.x;
    if (i < KK_) { cnt[i] = 0.f; ssum[i] = 0.f; acnt[i] = 0.f; }
    if (i == 0)  lacc[0] = 0.f;
    if (i < B_)  { bmax[i] = -2000000000; bmin[i] = 2000000000; }
}

// ---------------- one-time f32 -> f16 conversion of z and emb ----------------
__global__ void tohalf_kernel(const float* __restrict__ z, const float* __restrict__ emb,
                              _Float16* __restrict__ zh, _Float16* __restrict__ eh) {
    size_t i = (size_t)blockIdx.x * 256 + threadIdx.x;   // one float4 per thread
    const size_t nz4 = (size_t)M_ * E_ / 4;
    const size_t ne4 = (size_t)KK_ * E_ / 4;
    if (i < nz4) {
        float4 v = ((const float4*)z)[i];
        v4h h = { (_Float16)v.x, (_Float16)v.y, (_Float16)v.z, (_Float16)v.w };
        ((v4h*)zh)[i] = h;
    } else if (i < nz4 + ne4) {
        size_t j = i - nz4;
        float4 v = ((const float4*)emb)[j];
        v4h h = { (_Float16)v.x, (_Float16)v.y, (_Float16)v.z, (_Float16)v.w };
        ((v4h*)eh)[j] = h;
    }
}

// ---------------- row sums of squares (f32 originals) ----------------
__global__ void sumsq_kernel(const float* __restrict__ z, const float* __restrict__ emb,
                             float* __restrict__ z2, float* __restrict__ e2) {
    int i = blockIdx.x * 256 + threadIdx.x;
    if (i < M_) {
        const float4* p = (const float4*)(z + (size_t)i * E_);
        float s = 0.f;
        for (int q = 0; q < E_ / 4; ++q) {
            float4 v = p[q];
            s += v.x * v.x + v.y * v.y + v.z * v.z + v.w * v.w;
        }
        z2[i] = s;
    } else if (i < M_ + KK_) {
        int k = i - M_;
        const float4* p = (const float4*)(emb + (size_t)k * E_);
        float s = 0.f;
        for (int q = 0; q < E_ / 4; ++q) {
            float4 v = p[q];
            s += v.x * v.x + v.y * v.y + v.z * v.z + v.w * v.w;
        }
        e2[k] = s;
    }
}

// ---------------- WMMA distance GEMM: d = z2 + e2 - 2 * zh @ eh^T -------------
// Register-blocked: one wave computes a 64x64 output tile = 4x4 WMMA tiles.
// Operands are pre-converted f16; inner loop = 16 b128 loads + 16 WMMAs.
__global__ void __launch_bounds__(128)
dist_wmma(const _Float16* __restrict__ zh, const _Float16* __restrict__ eh,
          const float* __restrict__ z2, const float* __restrict__ e2,
          float* __restrict__ dmat) {
    const int wave = threadIdx.x >> 5;
    const int lane = threadIdx.x & 31;
    const int job  = blockIdx.x * 4 + wave;      // 0 .. 2303
    const int gm = job / 9;                       // 0 .. 255
    const int gn = job % 9;                       // 0 .. 8
    const int m0 = gm * 64;
    const int n0 = gn * 64;
    const int r16   = lane & 15;
    const int halfA = (lane < 16) ? 0 : 8;        // A per-lane K base (ISA layout)
    const int halfB = (lane < 16) ? 0 : 16;       // unused; B loads use halfA (same K order as A)

    const _Float16* zr[4];
    const _Float16* er[4];
    int ncol[4];
#pragma unroll
    for (int i = 0; i < 4; ++i) {
        zr[i] = zh + (size_t)(m0 + 16 * i + r16) * E_;
        int nr = n0 + 16 * i + r16;
        ncol[i] = nr;
        if (nr > KK_ - 1) nr = KK_ - 1;           // clamp reads; stores masked below
        er[i] = eh + (size_t)nr * E_;
    }

    v8f acc[4][4];
#pragma unroll
    for (int i = 0; i < 4; ++i)
#pragma unroll
        for (int j = 0; j < 4; ++j) acc[i][j] = (v8f){};

    for (int kk = 0; kk < E_; kk += 32) {
        // prefetch next K-chunk of the A stream (high-traffic stream)
        if (kk + 32 < E_) {
#pragma unroll
            for (int i = 0; i < 4; ++i)
                __builtin_prefetch(zr[i] + kk + 32 + halfA, 0, 3);
        }

        v16h af[4], bf[4];
#pragma unroll
        for (int i = 0; i < 4; ++i) {
            // A fragment: K = kk + {halfA..halfA+7, 16+halfA..16+halfA+7}
            v8h lo = *(const v8h*)(zr[i] + kk + halfA);
            v8h hi = *(const v8h*)(zr[i] + kk + 16 + halfA);
            af[i] = __builtin_shufflevector(lo, hi,
                     0, 1, 2, 3, 4, 5, 6, 7, 8, 9, 10, 11, 12, 13, 14, 15);
        }
#pragma unroll
        for (int j = 0; j < 4; ++j) {
            // B fragment: K = kk + {halfA..halfA+7, 16+halfA..16+halfA+7}
            v8h lo = *(const v8h*)(er[j] + kk + halfA);
            v8h hi = *(const v8h*)(er[j] + kk + 16 + halfA);
            bf[j] = __builtin_shufflevector(lo, hi,
                     0, 1, 2, 3, 4, 5, 6, 7, 8, 9, 10, 11, 12, 13, 14, 15);
        }
#pragma unroll
        for (int i = 0; i < 4; ++i)
#pragma unroll
            for (int j = 0; j < 4; ++j)
                acc[i][j] = __builtin_amdgcn_wmma_f32_16x16x32_f16(
                    false, af[i], false, bf[j], (short)0, acc[i][j], false, false);
    }

    // Epilogue: C/D layout — lane covers N = lane%16; VGPR r covers M = r + halfA.
#pragma unroll
    for (int j = 0; j < 4; ++j) {
        if (ncol[j] < KK_) {
            const float e2n = e2[ncol[j]];
#pragma unroll
            for (int i = 0; i < 4; ++i) {
                float accf[8];
#pragma unroll
                for (int r = 0; r < 8; ++r) accf[r] = acc[i][j][r];
#pragma unroll
                for (int r = 0; r < 8; ++r) {
                    int m = m0 + 16 * i + halfA + r;
                    dmat[(size_t)m * KK_ + ncol[j]] = z2[m] + e2n - 2.0f * accf[r];
                }
            }
        }
    }
}

// ---------------- greedy scan (1 wave per batch row, 4-step lookahead) -------
__global__ void scan_kernel(const float* __restrict__ dmat,
                            const unsigned char* __restrict__ zmask,
                            int* __restrict__ enc) {
    const int b = blockIdx.x;
    const int lane = threadIdx.x;   // 0..31

    // argmin over k of d[b, t=0, k] (first occurrence on ties)
    const float* row0 = dmat + (size_t)b * T_ * KK_;
    float best = 3.4e38f; int bi = 0;
    for (int k = lane; k < KK_; k += 32) {
        float v = row0[k];
        if (v < best) { best = v; bi = k; }
    }
    for (int off = 16; off > 0; off >>= 1) {
        float ob = __shfl_down(best, off);
        int   oi = __shfl_down(bi, off);
        if (ob < best || (ob == best && oi < bi)) { best = ob; bi = oi; }
    }
    bi = __shfl(bi, 0);
    int ind = bi > NE_ - 1 ? NE_ - 1 : bi;
    if (zmask[b]) ind = 0;
    float coe = 0.f;
    if (lane == 0) enc[b * T_ + 0] = ind;

    const int wi = lane >> 3;    // window row 0..3
    const int wj = lane & 7;     // window col 0..7
    for (int t0 = 1; t0 < T_; t0 += 4) {
        const int base = ind;    // uniform across wave
        int t = t0 + wi;
        int col = base + wj; if (col > KK_ - 1) col = KK_ - 1;
        float val = 0.f;
        if (t < T_) val = dmat[((size_t)b * T_ + t) * KK_ + col];
        for (int s = 0; s < 4; ++s) {
            int tt = t0 + s;
            if (tt >= T_) break;
            int indn = ind + 1; if (indn > NE_ - 1) indn = NE_ - 1;
            float dh = __shfl(val, s * 8 + (ind  - base));
            float dn = __shfl(val, s * 8 + (indn - base));
            bool stay = dh <= dn - coe;
            int indnew = stay ? ind : indn;
            coe = stay ? coe + PERS_STEP : 0.f;
            ind = indnew;
            if (lane == 0) enc[b * T_ + tt] = ind;
        }
    }
}

// ---------------- elastic update stages ----------------
__global__ void seg_count(const int* __restrict__ enc, const float* __restrict__ lc,
                          float* cnt, float* ssum) {
    int i = blockIdx.x * 256 + threadIdx.x;
    if (i < M_) {
        int k = enc[i];
        atomicAdd(&cnt[k], 1.0f);
        atomicAdd(&ssum[k], lc[i]);
    }
}

__global__ void mean_kernel(const float* cnt, const float* ssum, float* mean) {
    int k = blockIdx.x * 256 + threadIdx.x;
    if (k < KK_) mean[k] = (cnt[k] > 0.f) ? (ssum[k] / cnt[k]) : 0.f;
}

__global__ void anom_kernel(const int* __restrict__ enc, const float* __restrict__ lc,
                            const float* __restrict__ mean, float* acnt) {
    int i = blockIdx.x * 256 + threadIdx.x;
    if (i < M_) {
        int k = enc[i];
        float tol = 2.0f * mean[k];              // (1 + COE_LOSS_TOL) * mean
        if (tol < lc[i]) atomicAdd(&acnt[k], 1.0f);
    }
}

__global__ void bounds_kernel(const float* cnt, const float* acnt,
                              int* flinch, int* explore) {
    int k = blockIdx.x * 256 + threadIdx.x;
    if (k < KK_) {
        float c = cnt[k];
        float rate = (c > 0.f) ? (acnt[k] / c) : 0.f;
        int fl = (rate > FLINCH_B) ? 1 : 0;
        if (k == NE_ - 1) fl = 0;
        flinch[k] = fl;
        if (k == 0) explore[0] = 0;
        else {
            float cp = cnt[k - 1];
            float rp = (cp > 0.f) ? (acnt[k - 1] / cp) : 0.f;
            explore[k] = (rp < EXPLORE_B) ? 1 : 0;
        }
    }
}

__global__ void encfinal_kernel(const int* __restrict__ enc,
                                const int* __restrict__ flinch,
                                const int* __restrict__ explore,
                                const float* __restrict__ dmat,
                                int* __restrict__ encf, float* __restrict__ li,
                                float* __restrict__ out, int* bmax, int* bmin) {
    int i = blockIdx.x * 256 + threadIdx.x;
    if (i < M_) {
        int e = enc[i];
        int ef = e + flinch[e] - explore[e];
        ef = ef < 0 ? 0 : (ef > NE_ - 1 ? NE_ - 1 : ef);
        encf[i] = ef;
        li[i] = dmat[(size_t)i * KK_ + ef];
        out[ZQSZ + 1 + i] = (float)ef;           // enc output (as float)
        int b = i >> 9;                           // / T_
        atomicMax(&bmax[b], ef);
        atomicMin(&bmin[b], ef);
    }
}

// ---------------- z_q gather (straight-through forward value) ----------------
__global__ void zq_kernel(const float* __restrict__ emb, const int* __restrict__ encf,
                          float* __restrict__ out) {
    int i = blockIdx.x;                            // row 0..16383
    const float4* src = (const float4*)(emb + (size_t)encf[i] * E_);
    float4* dst = (float4*)(out + (size_t)i * E_);
    dst[threadIdx.x] = src[threadIdx.x];           // 128 threads x float4 = 512 floats
}

// ---------------- loss reduction ----------------
__global__ void loss_kernel(const float* __restrict__ dmat, const float* __restrict__ li,
                            float* lacc) {
    int m = blockIdx.x;
    float lim = li[m];
    const float* row = dmat + (size_t)m * KK_;
    float local = 0.f;
    for (int k = threadIdx.x; k < KK_; k += 256)
        local += fmaxf(lim - row[k] + EPS_, 0.f);
    for (int off = 16; off > 0; off >>= 1) local += __shfl_down(local, off);
    __shared__ float sm[8];
    if ((threadIdx.x & 31) == 0) sm[threadIdx.x >> 5] = local;
    __syncthreads();
    if (threadIdx.x < 8) {
        float v = sm[threadIdx.x];
        for (int off = 4; off > 0; off >>= 1) v += __shfl_down(v, off);
        if (threadIdx.x == 0) atomicAdd(lacc, v);
    }
}

// ---------------- finalize: loss scalar + v ----------------
__global__ void finalize_kernel(const float* lacc, const int* bmax, const int* bmin,
                                float* out) {
    int lane = threadIdx.x;
    int span = bmax[lane] - bmin[lane];
    for (int off = 16; off > 0; off >>= 1) {
        int o = __shfl_down(span, off);
        span = o > span ? o : span;
    }
    if (lane == 0) {
        out[ZQSZ] = lacc[0] * (1.0f + BETA_) / (float)((size_t)M_ * KK_);
        out[ZQSZ + 1 + M_] = (float)span;
    }
}

// ---------------- host launch ----------------
extern "C" void kernel_launch(void* const* d_in, const int* in_sizes, int n_in,
                              void* d_out, int out_size, void* d_ws, size_t ws_size,
                              hipStream_t stream) {
    const float* z    = (const float*)d_in[0];          // [B,T,E]
    const float* emb  = (const float*)d_in[1];          // [K,E]
    const float* lc   = (const float*)d_in[2];          // [B,T]
    const unsigned char* zmask = (const unsigned char*)d_in[3]; // [B] bool
    float* out = (float*)d_out;

    // workspace layout
    float* dmat   = (float*)d_ws;                       // M_*KK_ = 8,404,992 f32
    float* z2     = dmat + (size_t)M_ * KK_;            // 16384
    float* e2     = z2 + M_;                            // 513
    int*   enc    = (int*)(e2 + KK_);                   // 16384
    int*   encf   = enc + M_;                           // 16384
    float* cnt    = (float*)(encf + M_);                // 513
    float* ssum   = cnt + KK_;                          // 513
    float* mean   = ssum + KK_;                         // 513
    float* acnt   = mean + KK_;                         // 513
    int*   flinch = (int*)(acnt + KK_);                 // 513
    int*   explore= flinch + KK_;                       // 513
    float* li     = (float*)(explore + KK_);            // 16384
    float* lacc   = li + M_;                            // 1
    int*   bmax   = (int*)(lacc + 1);                   // 32
    int*   bmin   = bmax + B_;                          // 32
    _Float16* zh  = (_Float16*)(bmin + B_);             // M_*E_ halves (16 MB)
    _Float16* eh  = zh + (size_t)M_ * E_;               // KK_*E_ halves (0.5 MB)

    init_small<<<3, 256, 0, stream>>>(cnt, ssum, acnt, lacc, bmax, bmin);
    {
        const size_t n4 = ((size_t)M_ * E_ + (size_t)KK_ * E_) / 4;
        tohalf_kernel<<<(int)((n4 + 255) / 256), 256, 0, stream>>>(z, emb, zh, eh);
    }
    sumsq_kernel<<<(M_ + KK_ + 255) / 256, 256, 0, stream>>>(z, emb, z2, e2);
    dist_wmma<<<(256 * 9) / 4, 128, 0, stream>>>(zh, eh, z2, e2, dmat);
    scan_kernel<<<B_, 32, 0, stream>>>(dmat, zmask, enc);
    seg_count<<<(M_ + 255) / 256, 256, 0, stream>>>(enc, lc, cnt, ssum);
    mean_kernel<<<(KK_ + 255) / 256, 256, 0, stream>>>(cnt, ssum, mean);
    anom_kernel<<<(M_ + 255) / 256, 256, 0, stream>>>(enc, lc, mean, acnt);
    bounds_kernel<<<(KK_ + 255) / 256, 256, 0, stream>>>(cnt, acnt, flinch, explore);
    encfinal_kernel<<<(M_ + 255) / 256, 256, 0, stream>>>(enc, flinch, explore, dmat,
                                                          encf, li, out, bmax, bmin);
    zq_kernel<<<M_, 128, 0, stream>>>(emb, encf, out);
    loss_kernel<<<M_, 256, 0, stream>>>(dmat, li, lacc);
    finalize_kernel<<<1, 32, 0, stream>>>(lacc, bmax, bmin, out);
}
